// EncodeProcessDecode_37984690766437
// MI455X (gfx1250) — matrix operand order repaired and checked by reading, not compiled, not run on hardware
//
#include <hip/hip_runtime.h>
#include <cstdint>
#include <cstddef>

#define GN 50000
#define GE 800000
#define GD 128
#define GS 3

typedef __bf16 v16bf __attribute__((ext_vector_type(16)));
typedef float  v8f   __attribute__((ext_vector_type(8)));

union BFrag { v16bf v; uint4 q[2]; };

__device__ __forceinline__ v8f wmma_bf16(v16bf a, v16bf b, v8f c) {
  // D = A(16x32 bf16) * B(32x16 bf16) + C(16x16 f32)
  return __builtin_amdgcn_wmma_f32_16x16x32_bf16(false, a, false, b, (short)0, c,
                                                 false, false);
}

__device__ __forceinline__ uint16_t f2bf(float f) {
  uint32_t u = __float_as_uint(f);
  u += 0x7FFFu + ((u >> 16) & 1u);          // round-to-nearest-even
  return (uint16_t)(u >> 16);
}

__device__ __forceinline__ void load8(float* s, const float* __restrict__ p) {
  float4 a = *(const float4*)p;
  float4 b = *(const float4*)(p + 4);
  s[0]=a.x; s[1]=a.y; s[2]=a.z; s[3]=a.w;
  s[4]=b.x; s[5]=b.y; s[6]=b.z; s[7]=b.w;
}

// Fused  out = [LN]( (A @ W1 + b1) @ W2 + b2 )  for 128-wide layers.
// mode 0: A row m = f32 Aplain[m*Ksrc : +Ksrc] (zero-padded to Kpad), cvt->bf16
// mode 1: A row m = bf16 concat(x16[recv[m]], x16[send[m]], e16[m])  (Kpad=384)
// mode 2: A row m = bf16 concat(x16[m], agg16[m])                    (Kpad<=256)
// Wt1/Wt2 are bf16, pre-transposed: Wt[n*Kpad + k] = bf16(W[k][n]).
__global__ __launch_bounds__(128) void mlp_ln_wmma(
    const float* __restrict__ Aplain, int Ksrc, int Kpad, int mode,
    const int* __restrict__ recv, const int* __restrict__ send,
    const uint16_t* __restrict__ xbf, const uint16_t* __restrict__ ebf,
    const uint16_t* __restrict__ abf,
    const uint16_t* __restrict__ Wt1, const float* __restrict__ b1,
    int second, const uint16_t* __restrict__ Wt2, const float* __restrict__ b2,
    int do_ln, const float* __restrict__ g, const float* __restrict__ be,
    float* __restrict__ out, uint16_t* __restrict__ out_bf, int Mtiles)
{
  __shared__ float lds[4][16 * 132];         // per-wave 16x128 tile, padded stride
  const int lane   = threadIdx.x & 31;
  const int wid    = threadIdx.x >> 5;
  const int tile   = blockIdx.x * 4 + wid;
  if (tile >= Mtiles) return;                // wave-uniform: EXEC stays all-ones
  const int lanelo = lane & 15;
  const int hi     = lane >> 4;              // 0: K 0..7/16..23 ; 1: K 8..15/24..31
  const size_t arow = (size_t)tile * 16 + lanelo;

  const uint16_t* xr16 = nullptr; const uint16_t* xs16 = nullptr;
  const uint16_t* er16 = nullptr; const uint16_t* ag16 = nullptr;
  const float* abase = nullptr;
  if (mode == 1) {
    int r = recv[arow], s = send[arow];
    xr16 = xbf + (size_t)r * GD;
    xs16 = xbf + (size_t)s * GD;
    er16 = ebf + arow * GD;
  } else if (mode == 2) {
    xr16 = xbf + arow * GD;
    ag16 = abf + arow * GD;
  } else {
    abase = Aplain + arow * (size_t)Ksrc;
  }

  auto seg_bf = [&](int k0) -> uint4 {       // 8 consecutive bf16 (16B, aligned)
    const uint16_t* p;
    if (mode == 1) {
      if (k0 < 128)      p = xr16 + k0;
      else if (k0 < 256) p = xs16 + (k0 - 128);
      else               p = er16 + (k0 - 256);
    } else {
      p = (k0 < 128) ? (xr16 + k0) : (ag16 + (k0 - 128));
    }
    return *(const uint4*)p;
  };

  // ---------------- GEMM1: acc[t] = A @ W1, t = column tile (16 cols) ---------
  v8f acc[8] = {};
  for (int kc = 0; kc < Kpad; kc += 32) {
    BFrag fa;
    const int k0 = kc + hi * 8;
    if (mode == 0) {                         // f32 source, convert
      float s0[8], s1[8];
      if (k0 >= Ksrc) {
        #pragma unroll
        for (int i = 0; i < 8; ++i) s0[i] = 0.f;
      } else if (k0 + 8 <= Ksrc) {
        load8(s0, abase + k0);
      } else {
        #pragma unroll
        for (int i = 0; i < 8; ++i) { int k = k0 + i; s0[i] = (k < Ksrc) ? abase[k] : 0.f; }
      }
      const int k1 = k0 + 16;
      if (k1 >= Ksrc) {
        #pragma unroll
        for (int i = 0; i < 8; ++i) s1[i] = 0.f;
      } else if (k1 + 8 <= Ksrc) {
        load8(s1, abase + k1);
      } else {
        #pragma unroll
        for (int i = 0; i < 8; ++i) { int k = k1 + i; s1[i] = (k < Ksrc) ? abase[k] : 0.f; }
      }
      #pragma unroll
      for (int i = 0; i < 8; ++i) {
        fa.v[i]     = (__bf16)s0[i];
        fa.v[8 + i] = (__bf16)s1[i];
      }
    } else {                                 // bf16 mirror gather: 2x b128 loads
      fa.q[0] = seg_bf(k0);
      fa.q[1] = seg_bf(k0 + 16);
    }
    const uint16_t* wb = Wt1 + (size_t)lanelo * Kpad + kc + hi * 8;
    #pragma unroll
    for (int t = 0; t < 8; ++t) {
      BFrag fb;
      const uint16_t* wp = wb + (size_t)t * 16 * Kpad;
      fb.q[0] = *(const uint4*)(wp);
      fb.q[1] = *(const uint4*)(wp + 16);
      acc[t] = wmma_bf16(fa.v, fb.v, acc[t]);
    }
  }
  // bias1
  #pragma unroll
  for (int t = 0; t < 8; ++t) {
    float bc = b1[16 * t + lanelo];
    #pragma unroll
    for (int r = 0; r < 8; ++r) acc[t][r] += bc;
  }

  if (!second) {   // single-layer mode (decoder hidden): write h = A@W1+b1
    #pragma unroll
    for (int t = 0; t < 8; ++t)
      #pragma unroll
      for (int r = 0; r < 8; ++r)
        out[((size_t)tile * 16 + 8 * hi + r) * GD + 16 * t + lanelo] = acc[t][r];
    return;
  }

  // ------------- hidden tile -> LDS -> re-fragment as bf16 A ------------------
  float* tl = lds[wid];
  #pragma unroll
  for (int t = 0; t < 8; ++t)
    #pragma unroll
    for (int r = 0; r < 8; ++r)
      tl[(r + 8 * hi) * 132 + 16 * t + lanelo] = acc[t][r];
  asm volatile("s_wait_dscnt 0" ::: "memory");   // same-wave LDS RAW fence

  // ---------------- GEMM2: acc2 = h @ W2 (K = 128) ---------------------------
  v8f acc2[8] = {};
  for (int kc = 0; kc < GD; kc += 32) {
    BFrag fa;
    {
      float s[16];
      const float* lr = tl + lanelo * 132 + kc + hi * 8;
      *(float4*)(s)      = *(const float4*)(lr);
      *(float4*)(s + 4)  = *(const float4*)(lr + 4);
      *(float4*)(s + 8)  = *(const float4*)(lr + 16);
      *(float4*)(s + 12) = *(const float4*)(lr + 20);
      #pragma unroll
      for (int i = 0; i < 16; ++i) fa.v[i] = (__bf16)s[i];
    }
    const uint16_t* wb = Wt2 + (size_t)lanelo * GD + kc + hi * 8;
    #pragma unroll
    for (int t = 0; t < 8; ++t) {
      BFrag fb;
      const uint16_t* wp = wb + (size_t)t * 16 * GD;
      fb.q[0] = *(const uint4*)(wp);
      fb.q[1] = *(const uint4*)(wp + 16);
      acc2[t] = wmma_bf16(fa.v, fb.v, acc2[t]);
    }
  }
  #pragma unroll
  for (int t = 0; t < 8; ++t) {
    float bc = b2[16 * t + lanelo];
    #pragma unroll
    for (int r = 0; r < 8; ++r) acc2[t][r] += bc;
  }

  // ---------------- LayerNorm across the 128 columns (in-register) -----------
  if (do_ln) {
    float sm[8], sq[8];
    #pragma unroll
    for (int r = 0; r < 8; ++r) { sm[r] = 0.f; sq[r] = 0.f; }
    #pragma unroll
    for (int t = 0; t < 8; ++t)
      #pragma unroll
      for (int r = 0; r < 8; ++r) { float v = acc2[t][r]; sm[r] += v; sq[r] += v * v; }
    #pragma unroll
    for (int off = 1; off < 16; off <<= 1) {
      #pragma unroll
      for (int r = 0; r < 8; ++r) {
        sm[r] += __shfl_xor(sm[r], off, 32);
        sq[r] += __shfl_xor(sq[r], off, 32);
      }
    }
    float mu[8], rs[8];
    #pragma unroll
    for (int r = 0; r < 8; ++r) {
      mu[r] = sm[r] * (1.0f / 128.0f);
      float var = sq[r] * (1.0f / 128.0f) - mu[r] * mu[r];
      rs[r] = rsqrtf(var + 1e-5f);
    }
    #pragma unroll
    for (int t = 0; t < 8; ++t) {
      float gc = g[16 * t + lanelo], bc = be[16 * t + lanelo];
      #pragma unroll
      for (int r = 0; r < 8; ++r)
        acc2[t][r] = (acc2[t][r] - mu[r]) * rs[r] * gc + bc;
    }
  }

  #pragma unroll
  for (int t = 0; t < 8; ++t)
    #pragma unroll
    for (int r = 0; r < 8; ++r) {
      size_t o = ((size_t)tile * 16 + 8 * hi + r) * GD + 16 * t + lanelo;
      float v = acc2[t][r];
      out[o] = v;
      if (out_bf) out_bf[o] = f2bf(v);       // keep bf16 mirror coherent
    }
}

// ---- weight transpose + f32->bf16 convert: Wt[n*Kpad+k] = bf16(W[k*Nn+n]) ----
__global__ void wt_conv_kernel(const float* __restrict__ W, uint16_t* __restrict__ Wt,
                               int K, int Kpad, int Nn) {
  int idx = blockIdx.x * blockDim.x + threadIdx.x;
  if (idx >= Nn * Kpad) return;
  int n = idx / Kpad, k = idx % Kpad;
  float v = (k < K) ? W[(size_t)k * Nn + n] : 0.f;
  Wt[idx] = f2bf(v);
}

__global__ void zero_kernel(float* __restrict__ p, int n) {
  int i = blockIdx.x * blockDim.x + threadIdx.x;
  if (i < n) p[i] = 0.f;
}

__global__ void count_kernel(const int* __restrict__ recv, float* __restrict__ cnt, int E) {
  int e = blockIdx.x * blockDim.x + threadIdx.x;
  if (e < E) atomicAdd(&cnt[recv[e]], 1.0f);
}

__global__ void scatter_add_kernel(const float* __restrict__ vals, const int* __restrict__ recv,
                                   float* __restrict__ sums, int E) {
  int tid = blockIdx.x * blockDim.x + threadIdx.x;
  int e = tid >> 5, q = tid & 31;
  if (e >= E) return;
  int r = recv[e];
  float4 v = ((const float4*)(vals + (size_t)e * GD))[q];
  float* d = sums + (size_t)r * GD + q * 4;
  atomicAdd(d + 0, v.x); atomicAdd(d + 1, v.y);
  atomicAdd(d + 2, v.z); atomicAdd(d + 3, v.w);
}

// sums /= max(cnt,1) and emit bf16 mirror
__global__ void mean_div_mirror_kernel(float* __restrict__ sums, const float* __restrict__ cnt,
                                       uint16_t* __restrict__ mir, int total) {
  int i = blockIdx.x * blockDim.x + threadIdx.x;
  if (i >= total) return;
  float v = sums[i] / fmaxf(cnt[i >> 7], 1.0f);
  sums[i] = v;
  mir[i] = f2bf(v);
}

// dst += src (f32 master) and refresh bf16 mirror
__global__ void axpy_mirror_kernel(float* __restrict__ dst, const float* __restrict__ src,
                                   uint16_t* __restrict__ mir, int n) {
  int i = blockIdx.x * blockDim.x + threadIdx.x;
  if (i >= n) return;
  float v = dst[i] + src[i];
  dst[i] = v;
  mir[i] = f2bf(v);
}

__global__ void decoder_out_kernel(const float* __restrict__ h, const float* __restrict__ W2,
                                   const float* __restrict__ b2, float* __restrict__ out, int M) {
  int m = blockIdx.x * blockDim.x + threadIdx.x;
  if (m >= M) return;
  float a0 = b2[0], a1 = b2[1], a2 = b2[2];
  const float* hr = h + (size_t)m * GD;
  #pragma unroll 4
  for (int d = 0; d < GD; ++d) {
    float v = hr[d];
    a0 += v * W2[d * 3 + 0];
    a1 += v * W2[d * 3 + 1];
    a2 += v * W2[d * 3 + 2];
  }
  out[(size_t)m * 3 + 0] = a0;
  out[(size_t)m * 3 + 1] = a1;
  out[(size_t)m * 3 + 2] = a2;
}

static inline int cdiv(long long a, long long b) { return (int)((a + b - 1) / b); }

extern "C" void kernel_launch(void* const* d_in, const int* in_sizes, int n_in,
                              void* d_out, int out_size, void* d_ws, size_t ws_size,
                              hipStream_t stream) {
  const float* node_feat = (const float*)d_in[0];
  const float* edge_feat = (const float*)d_in[1];
  const int*   edge_index = (const int*)d_in[2];
  const int* recv = edge_index;
  const int* send = edge_index + GE;
  const float* en_W1 = (const float*)d_in[3];  const float* en_b1 = (const float*)d_in[4];
  const float* en_W2 = (const float*)d_in[5];  const float* en_b2 = (const float*)d_in[6];
  const float* en_g  = (const float*)d_in[7];  const float* en_be = (const float*)d_in[8];
  const float* ee_W1 = (const float*)d_in[9];  const float* ee_b1 = (const float*)d_in[10];
  const float* ee_W2 = (const float*)d_in[11]; const float* ee_b2 = (const float*)d_in[12];
  const float* ee_g  = (const float*)d_in[13]; const float* ee_be = (const float*)d_in[14];
  const float* pe_W1 = (const float*)d_in[15]; const float* pe_b1 = (const float*)d_in[16];
  const float* pe_W2 = (const float*)d_in[17]; const float* pe_b2 = (const float*)d_in[18];
  const float* pe_g  = (const float*)d_in[19]; const float* pe_be = (const float*)d_in[20];
  const float* pn_W1 = (const float*)d_in[21]; const float* pn_b1 = (const float*)d_in[22];
  const float* pn_W2 = (const float*)d_in[23]; const float* pn_b2 = (const float*)d_in[24];
  const float* pn_g  = (const float*)d_in[25]; const float* pn_be = (const float*)d_in[26];
  const float* de_W1 = (const float*)d_in[27]; const float* de_b1 = (const float*)d_in[28];
  const float* de_W2 = (const float*)d_in[29]; const float* de_b2 = (const float*)d_in[30];

  // -------- workspace carve --------
  char* cur = (char*)d_ws;
  auto carve = [&](size_t bytes) -> void* {
    void* p = (void*)cur;
    cur += (bytes + 255) & ~(size_t)255;
    return p;
  };
  float* xb   = (float*)carve((size_t)GN * GD * 4);   // x master (f32)
  float* eb   = (float*)carve((size_t)GE * GD * 4);   // e master (f32)
  float* enew = (float*)carve((size_t)GE * GD * 4);   // e_new
  float* aggs = (float*)carve((size_t)GN * GD * 4);   // scatter sums / means
  float* nh1  = (float*)carve((size_t)GN * GD * 4);   // n_new / decoder hidden
  float* cnt  = (float*)carve((size_t)GN * 4);
  uint16_t* x16 = (uint16_t*)carve((size_t)GN * GD * 2);   // bf16 mirrors
  uint16_t* e16 = (uint16_t*)carve((size_t)GE * GD * 2);
  uint16_t* a16 = (uint16_t*)carve((size_t)GN * GD * 2);
  uint16_t* wt_en1 = (uint16_t*)carve((size_t)128 * 32 * 2);
  uint16_t* wt_en2 = (uint16_t*)carve((size_t)128 * 128 * 2);
  uint16_t* wt_ee1 = (uint16_t*)carve((size_t)128 * 32 * 2);
  uint16_t* wt_ee2 = (uint16_t*)carve((size_t)128 * 128 * 2);
  uint16_t *wt_pe1[GS], *wt_pe2[GS], *wt_pn1[GS], *wt_pn2[GS];
  for (int s = 0; s < GS; ++s) {
    wt_pe1[s] = (uint16_t*)carve((size_t)128 * 384 * 2);
    wt_pe2[s] = (uint16_t*)carve((size_t)128 * 128 * 2);
    wt_pn1[s] = (uint16_t*)carve((size_t)128 * 256 * 2);
    wt_pn2[s] = (uint16_t*)carve((size_t)128 * 128 * 2);
  }
  uint16_t* wt_de1 = (uint16_t*)carve((size_t)128 * 128 * 2);
  if ((size_t)(cur - (char*)d_ws) > ws_size) return;  // insufficient scratch

  const int TB = 256;
  auto wconv = [&](const float* W, uint16_t* Wt, int K, int Kpad) {
    wt_conv_kernel<<<cdiv(128 * Kpad, TB), TB, 0, stream>>>(W, Wt, K, Kpad, 128);
  };
  wconv(en_W1, wt_en1, 32, 32);
  wconv(en_W2, wt_en2, 128, 128);
  wconv(ee_W1, wt_ee1, 4, 32);
  wconv(ee_W2, wt_ee2, 128, 128);
  for (int s = 0; s < GS; ++s) {
    wconv(pe_W1 + (size_t)s * 384 * 128, wt_pe1[s], 384, 384);
    wconv(pe_W2 + (size_t)s * 128 * 128, wt_pe2[s], 128, 128);
    wconv(pn_W1 + (size_t)s * 256 * 128, wt_pn1[s], 256, 256);
    wconv(pn_W2 + (size_t)s * 128 * 128, wt_pn2[s], 128, 128);
  }
  wconv(de_W1, wt_de1, 128, 128);

  const int NT = GN / 16;   // 3125 node tiles
  const int ET = GE / 16;   // 50000 edge tiles
  dim3 mblk(128);

  // encode nodes: x = MLP_LN(node_feat)  (mirror refreshed after aggregation)
  mlp_ln_wmma<<<cdiv(NT, 4), mblk, 0, stream>>>(
      node_feat, 32, 32, 0, nullptr, nullptr, nullptr, nullptr, nullptr,
      wt_en1, en_b1, 1, wt_en2, en_b2, 1, en_g, en_be, xb, nullptr, NT);
  // encode edges: e = MLP_LN(edge_feat)  (+ e16 mirror)
  mlp_ln_wmma<<<cdiv(ET, 4), mblk, 0, stream>>>(
      edge_feat, 4, 32, 0, nullptr, nullptr, nullptr, nullptr, nullptr,
      wt_ee1, ee_b1, 1, wt_ee2, ee_b2, 1, ee_g, ee_be, eb, e16, ET);

  // degree counts (recv is constant across steps)
  zero_kernel<<<cdiv(GN, TB), TB, 0, stream>>>(cnt, GN);
  count_kernel<<<cdiv(GE, TB), TB, 0, stream>>>(recv, cnt, GE);

  // x += scatter_mean(e, recv)   (also writes x16 mirror)
  zero_kernel<<<cdiv(GN * GD, TB), TB, 0, stream>>>(aggs, GN * GD);
  scatter_add_kernel<<<cdiv((long long)GE * 32, TB), TB, 0, stream>>>(eb, recv, aggs, GE);
  mean_div_mirror_kernel<<<cdiv(GN * GD, TB), TB, 0, stream>>>(aggs, cnt, a16, GN * GD);
  axpy_mirror_kernel<<<cdiv(GN * GD, TB), TB, 0, stream>>>(xb, aggs, x16, GN * GD);

  // processor steps
  for (int s = 0; s < GS; ++s) {
    // e_new = MLP_LN(concat(x[recv], x[send], e))  -- bf16 gathers
    mlp_ln_wmma<<<cdiv(ET, 4), mblk, 0, stream>>>(
        nullptr, 384, 384, 1, recv, send, x16, e16, nullptr,
        wt_pe1[s], pe_b1 + s * GD, 1, wt_pe2[s], pe_b2 + s * GD,
        1, pe_g + s * GD, pe_be + s * GD, enew, nullptr, ET);
    // agg = scatter_mean(e_new, recv)  (+ a16 mirror)
    zero_kernel<<<cdiv(GN * GD, TB), TB, 0, stream>>>(aggs, GN * GD);
    scatter_add_kernel<<<cdiv((long long)GE * 32, TB), TB, 0, stream>>>(enew, recv, aggs, GE);
    mean_div_mirror_kernel<<<cdiv(GN * GD, TB), TB, 0, stream>>>(aggs, cnt, a16, GN * GD);
    // n_new = MLP_LN(concat(x, agg))  -- bf16 reads
    mlp_ln_wmma<<<cdiv(NT, 4), mblk, 0, stream>>>(
        nullptr, 256, 256, 2, nullptr, nullptr, x16, nullptr, a16,
        wt_pn1[s], pn_b1 + s * GD, 1, wt_pn2[s], pn_b2 + s * GD,
        1, pn_g + s * GD, pn_be + s * GD, nh1, nullptr, NT);
    // residuals on f32 masters, refresh mirrors
    axpy_mirror_kernel<<<cdiv(GN * GD, TB), TB, 0, stream>>>(xb, nh1, x16, GN * GD);
    axpy_mirror_kernel<<<cdiv(GE * GD, TB), TB, 0, stream>>>(eb, enew, e16, GE * GD);
  }

  // decode: h = x @ de_W1 + de_b1 (single layer, no LN, bf16 x mirror input)
  mlp_ln_wmma<<<cdiv(NT, 4), mblk, 0, stream>>>(
      nullptr, 128, 128, 2, nullptr, nullptr, x16, nullptr, nullptr,
      wt_de1, de_b1, 0, nullptr, nullptr, 0, nullptr, nullptr, nh1, nullptr, NT);
  decoder_out_kernel<<<cdiv(GN, TB), TB, 0, stream>>>(nh1, de_W2, de_b2, (float*)d_out, GN);
}
